// Linear_73177652789981
// MI455X (gfx1250) — 8-bit matrix operand order repaired and checked by reading, compiled, not run on hardware
//
#include <hip/hip_runtime.h>
#include <stdint.h>


typedef int v8i __attribute__((ext_vector_type(8)));

#define NROWS 32768
#define DIN   1024
#define DOUT  1024

#define BM 256
#define BN 128
#define BK 64

__device__ __forceinline__ void async_load_b128(const void* gsrc, void* ldst) {
    uint32_t loff  = (uint32_t)(uintptr_t)ldst;
    uint64_t gaddr = (uint64_t)(uintptr_t)gsrc;
    asm volatile("global_load_async_to_lds_b128 %0, %1, off"
                 :: "v"(loff), "v"(gaddr) : "memory");
}

__device__ __forceinline__ void wait_async0() {
    asm volatile("s_wait_asynccnt 0" ::: "memory");
}

__global__ __launch_bounds__(256) void pack_i32_to_i8(const int4* __restrict__ src,
                                                      int* __restrict__ dst, int n4) {
    int i = blockIdx.x * 256 + threadIdx.x;
    if (i < n4) {
        int4 v = src[i];
        dst[i] = (v.x & 0xFF) | ((v.y & 0xFF) << 8) | ((v.z & 0xFF) << 16) | (v.w << 24);
    }
}

__global__ __launch_bounds__(256) void qgemm_wmma_iu8(
    const unsigned char* __restrict__ xq,
    const unsigned char* __restrict__ wq,
    const int*   __restrict__ bias,
    const float* __restrict__ act_scale,
    const float* __restrict__ wt_scale,
    const float* __restrict__ out_scale,
    float* __restrict__ out)
{
    __shared__ __align__(16) unsigned char lA[2][BM * BK];
    __shared__ __align__(16) unsigned char lB[2][BN * BK];

    const int tid  = threadIdx.x;
    const int lane = tid & 31;
    const int wid  = tid >> 5;
    const int wm   = wid >> 1;
    const int wn   = wid & 1;

    const int mBase = blockIdx.y * BM;
    const int nBase = blockIdx.x * BN;

    v8i acc[4][4];
#pragma unroll
    for (int m = 0; m < 4; ++m)
#pragma unroll
        for (int n = 0; n < 4; ++n)
#pragma unroll
            for (int r = 0; r < 8; ++r) acc[m][n][r] = 0;

    const int lrow = tid >> 2;
    const int loff = (tid & 3) * 16;

    const unsigned char* gA = xq + (size_t)(mBase + lrow) * DIN + loff;
    const unsigned char* gB = wq + (size_t)(nBase + lrow) * DIN + loff;

    const int aKb = (lane >> 4) * 16;
    const int bKb = (lane >> 4) * 16;
    const int hlf = lane & 15;

#define STAGE(pp, kk)                                                          \
    do {                                                                       \
        unsigned char* dA = &lA[pp][lrow * BK + loff];                         \
        unsigned char* dB = &lB[pp][lrow * BK + loff];                         \
        _Pragma("unroll")                                                      \
        for (int r = 0; r < 4; ++r)          \
            async_load_b128(gA + (size_t)r * 64 * DIN + (kk),                  \
                            dA + r * 64 * BK);                                 \
        _Pragma("unroll")                                                      \
        for (int r = 0; r < 2; ++r)          \
            async_load_b128(gB + (size_t)r * 64 * DIN + (kk),                  \
                            dB + r * 64 * BK);                                 \
    } while (0)

    STAGE(0, 0);
    wait_async0();
    __syncthreads();

    int p = 0;
    for (int k0 = 0; k0 < DIN; k0 += BK) {
        if (k0 + BK < DIN) STAGE(p ^ 1, k0 + BK);
        if (k0 + 2 * BK < DIN) {
            __builtin_prefetch(gA + k0 + 2 * BK, 0, 3);
            __builtin_prefetch(gB + k0 + 2 * BK, 0, 3);
        }

        v8i af[4], bf[4];
#pragma unroll
        for (int m = 0; m < 4; ++m) {
            const unsigned char* pa =
                &lA[p][(wm * 64 + m * 16 + hlf) * BK + aKb];
            int4 t0 = *(const int4*)(pa);
            int4 t1 = *(const int4*)(pa + 32);
            af[m][0] = t0.x; af[m][1] = t0.y; af[m][2] = t0.z; af[m][3] = t0.w;
            af[m][4] = t1.x; af[m][5] = t1.y; af[m][6] = t1.z; af[m][7] = t1.w;
        }
#pragma unroll
        for (int n = 0; n < 4; ++n) {
            const unsigned char* pb =
                &lB[p][(wn * 64 + n * 16 + hlf) * BK + bKb];
            int4 t0 = *(const int4*)(pb);
            int4 t1 = *(const int4*)(pb + 32);
            bf[n][0] = t0.x; bf[n][1] = t0.y; bf[n][2] = t0.z; bf[n][3] = t0.w;
            bf[n][4] = t1.x; bf[n][5] = t1.y; bf[n][6] = t1.z; bf[n][7] = t1.w;
        }

#pragma unroll
        for (int m = 0; m < 4; ++m)
#pragma unroll
            for (int n = 0; n < 4; ++n)
                acc[m][n] = __builtin_amdgcn_wmma_i32_16x16x64_iu8(
                     true, af[m],
                     true, bf[n],
                    acc[m][n],  false,  false);

        wait_async0();
        __syncthreads();
        p ^= 1;
    }
#undef STAGE

    const float sAct = act_scale[0];
    const float rOut = 1.0f / out_scale[0];

#pragma unroll
    for (int n = 0; n < 4; ++n) {
        const int ng = nBase + wn * 64 + n * 16 + hlf;
        const float sc = sAct * wt_scale[ng] * rOut;
        const int   bs = bias[ng];
#pragma unroll
        for (int m = 0; m < 4; ++m) {
            const int mg = mBase + wm * 64 + m * 16 + ((lane >> 4) * 8);
#pragma unroll
            for (int r = 0; r < 8; ++r) {
                float v = (float)(acc[m][n][r] + bs) * sc;
                v = fminf(fmaxf(v, -128.0f), 127.0f);
                out[(size_t)(mg + r) * DOUT + ng] = v;
            }
        }
    }
}

extern "C" void kernel_launch(void* const* d_in, const int* in_sizes, int n_in,
                              void* d_out, int out_size, void* d_ws, size_t ws_size,
                              hipStream_t stream) {
    const int*   x         = (const int*)d_in[0];
    const int*   w         = (const int*)d_in[1];
    const int*   bias      = (const int*)d_in[2];
    const float* act_scale = (const float*)d_in[3];
    const float* wt_scale  = (const float*)d_in[4];
    const float* out_scale = (const float*)d_in[5];
    float* out = (float*)d_out;

    unsigned char* xq = (unsigned char*)d_ws;
    unsigned char* wq = xq + (size_t)NROWS * DIN;

    const int n4x = NROWS * DIN / 4;
    const int n4w = DOUT * DIN / 4;
    pack_i32_to_i8<<<(n4x + 255) / 256, 256, 0, stream>>>((const int4*)x, (int*)xq, n4x);
    pack_i32_to_i8<<<(n4w + 255) / 256, 256, 0, stream>>>((const int4*)w, (int*)wq, n4w);

    dim3 grid(DOUT / BN, NROWS / BM);
    qgemm_wmma_iu8<<<grid, 256, 0, stream>>>(xq, wq, bias, act_scale, wt_scale,
                                             out_scale, out);
}
